// BatteryMoEFlattenIntraCycleMoELayer_70042326663576
// MI455X (gfx1250) — matrix operand order repaired and checked by reading, not compiled, not run on hardware
//
#include <hip/hip_runtime.h>
#include <hip/hip_bf16.h>

typedef __attribute__((ext_vector_type(16))) __bf16 v16bf;
typedef __attribute__((ext_vector_type(8)))  __bf16 v8bf;
typedef __attribute__((ext_vector_type(8)))  float  v8f;

#define B_    128
#define L_    128
#define F_    1536
#define E_    16
#define D_    768
#define KSTEP 64            // K elements staged per iteration (2 WMMA chunks)
#define KPAD  (KSTEP + 8)   // LDS row padding -> conflict-free B-frag reads
#define BG    16            // batch samples per block (weight-load amortization)
#define LT    4             // 16-row L-tiles per wave
#define NTHR  512           // 16 wave32

// ---------------------------------------------------------------------------
// Kernel 1: masked, renormalized softmax gates g[b,e]  (B=128 threads)
// ---------------------------------------------------------------------------
__global__ __launch_bounds__(128) void gates_kernel(
    const float* __restrict__ logits,   // [B, E]
    const int*   __restrict__ masks,    // [B, E]
    float*       __restrict__ g)        // [B, E] out (workspace)
{
    const int b = threadIdx.x;
    float l[E_];
    float m = -3.4e38f;
#pragma unroll
    for (int e = 0; e < E_; ++e) { l[e] = logits[b * E_ + e]; m = fmaxf(m, l[e]); }
    float s = 0.f;
#pragma unroll
    for (int e = 0; e < E_; ++e) { l[e] = __expf(l[e] - m); s += l[e]; }
    const float inv = 1.f / s;
    float gsum = 0.f;
#pragma unroll
    for (int e = 0; e < E_; ++e) {
        float p = l[e] * inv * (masks[b * E_ + e] == 1 ? 1.f : 0.f);
        l[e] = p; gsum += p;
    }
    const float r = 1.f / (gsum + 1e-9f);
#pragma unroll
    for (int e = 0; e < E_; ++e) g[b * E_ + e] = l[e] * r;
}

static __device__ __forceinline__ unsigned pk_bf16(float a, float b) {
    union { __bf16 h[2]; unsigned u; } p;
    p.h[0] = (__bf16)a; p.h[1] = (__bf16)b;
    return p.u;
}

static __device__ __forceinline__ v16bf lds_b_frag(const __bf16* p) {
    // 16B-aligned LDS reads -> two ds_load_b128
    v8bf lo = *(const v8bf*)(p);
    v8bf hi = *(const v8bf*)(p + 8);
    return __builtin_shufflevector(lo, hi, 0,1,2,3,4,5,6,7,8,9,10,11,12,13,14,15);
}

// ---------------------------------------------------------------------------
// Kernel 2: fused MoE combine + dual GEMM via v_wmma_f32_16x16x32_bf16.
// Grid (48, 8, 2); block owns (16-wide d-tile, 16 b's, 64 L-rows).
// Wave w handles b = b_base + w, four 16-row L-tiles.
// ---------------------------------------------------------------------------
__global__ __launch_bounds__(NTHR) void moe_wmma_kernel(
    const float* __restrict__ x,      // [B, L, F]
    const float* __restrict__ eW,     // [E, D, F]
    const float* __restrict__ eb,     // [E, D]
    const float* __restrict__ gW,     // [2, D, F]
    const float* __restrict__ gb,     // [2, D]
    const float* __restrict__ gates,  // [B, E] (workspace)
    float*       __restrict__ out)    // [B, L, D]
{
    __shared__ __bf16 wex[BG][16][KPAD];   // W_eff tiles, one per b   (~36 KB)
    __shared__ __bf16 wgn[16][KPAD];       // W_gen tile               (~2 KB)
    __shared__ float  g_s[BG][E_];         // gates for this b-group   (1 KB)

    const int tid    = threadIdx.x;
    const int d0     = blockIdx.x * 16;
    const int b_base = blockIdx.y * BG;
    const int lbase  = blockIdx.z * (16 * LT);

    const int lane  = tid & 31;
    const int wave  = tid >> 5;           // local b index (0..15)
    const int b     = b_base + wave;
    const int n     = lane & 15;          // A row m / B col n / C col n
    const int khalf = lane >> 4;

    // staging slot: thread owns (d = sd, k-pair = sk2) for ALL 16 b's
    const int sd  = tid >> 5;             // 0..15
    const int sk2 = tid & 31;             // 0..31 (pairs of k within KSTEP)

    if (tid < BG * E_) ((float*)g_s)[tid] = gates[b_base * E_ + tid];
    __syncthreads();

    v8f acc_e[LT], acc_g[LT];
#pragma unroll
    for (int t = 0; t < LT; ++t) { acc_e[t] = (v8f){}; acc_g[t] = (v8f){}; }

    for (int kb = 0; kb < F_; kb += KSTEP) {
        // ================= stage: W_eff (all 16 b) + W_gen, bf16, in LDS ====
        {
            const float* wp = eW + (size_t)(d0 + sd) * F_ + kb + 2 * sk2;
            float2 w[E_];                                  // load each weight ONCE
#pragma unroll
            for (int e = 0; e < E_; ++e)
                w[e] = *(const float2*)(wp + (size_t)e * ((size_t)D_ * F_));
            if (kb + KSTEP < F_)
                __builtin_prefetch(wp + KSTEP, 0, 1);      // next chunk, expert 0

#pragma unroll
            for (int bb = 0; bb < BG; ++bb) {              // reuse across 16 b's
                float a0 = 0.f, a1 = 0.f;
#pragma unroll
                for (int e = 0; e < E_; ++e) {
                    const float gg = g_s[bb][e];           // LDS broadcast (uniform addr)
                    a0 = fmaf(gg, w[e].x, a0);
                    a1 = fmaf(gg, w[e].y, a1);
                }
                *(unsigned*)&wex[bb][sd][2 * sk2] = pk_bf16(a0, a1);
            }
            const float* gp = gW + (size_t)(d0 + sd) * F_ + kb + 2 * sk2;
            const float q0 = gp[0] + gp[(size_t)D_ * F_];
            const float q1 = gp[1] + gp[(size_t)D_ * F_ + 1];
            *(unsigned*)&wgn[sd][2 * sk2] = pk_bf16(q0, q1);
        }
        __syncthreads();

        // ================= consume: 2 K=32 chunks x LT l-tiles x 2 paths ====
#pragma unroll
        for (int c = 0; c < 2; ++c) {
            const v16bf be = lds_b_frag(&wex[wave][n][c * 32 + khalf * 16]);
            const v16bf bg = lds_b_frag(&wgn[n][c * 32 + khalf * 16]);
#pragma unroll
            for (int tt = 0; tt < LT; ++tt) {
                // A layout matches B frag: lane holds K in [16*khalf,+16)
                const float* xrow =
                    x + ((size_t)b * L_ + lbase + tt * 16 + n) * F_ + kb + c * 32;
                const float4 v0 = *(const float4*)(xrow + khalf * 16);
                const float4 v1 = *(const float4*)(xrow + khalf * 16 + 4);
                const float4 v2 = *(const float4*)(xrow + khalf * 16 + 8);
                const float4 v3 = *(const float4*)(xrow + khalf * 16 + 12);
                v16bf af;
                af[0]  = (__bf16)v0.x; af[1]  = (__bf16)v0.y;
                af[2]  = (__bf16)v0.z; af[3]  = (__bf16)v0.w;
                af[4]  = (__bf16)v1.x; af[5]  = (__bf16)v1.y;
                af[6]  = (__bf16)v1.z; af[7]  = (__bf16)v1.w;
                af[8]  = (__bf16)v2.x; af[9]  = (__bf16)v2.y;
                af[10] = (__bf16)v2.z; af[11] = (__bf16)v2.w;
                af[12] = (__bf16)v3.x; af[13] = (__bf16)v3.y;
                af[14] = (__bf16)v3.z; af[15] = (__bf16)v3.w;

                acc_e[tt] = __builtin_amdgcn_wmma_f32_16x16x32_bf16(
                                false, af, false, be, (short)0, acc_e[tt], false, false);
                acc_g[tt] = __builtin_amdgcn_wmma_f32_16x16x32_bf16(
                                false, af, false, bg, (short)0, acc_g[tt], false, false);
            }
        }
        __syncthreads();   // LDS tiles consumed; safe to overwrite next iter
    }

    // ================= epilogue: biases, bf16 rounding, f32 store ==========
    float beff = 0.f;
#pragma unroll
    for (int e = 0; e < E_; ++e)
        beff = fmaf(g_s[wave][e], eb[e * D_ + d0 + n], beff);
    const float bgen = gb[d0 + n] + gb[D_ + d0 + n];

#pragma unroll
    for (int tt = 0; tt < LT; ++tt) {
#pragma unroll
        for (int r = 0; r < 8; ++r) {
            const int l = lbase + tt * 16 + khalf * 8 + r;   // C layout: M = r + 8*(lane/16)
            const float comb_bf = (float)(__bf16)(acc_e[tt][r] + beff);  // bf16 round
            const float gen     = acc_g[tt][r] + bgen;                   // stays f32
            out[((size_t)b * L_ + l) * D_ + d0 + n] = comb_bf + gen;
        }
    }
}

// ---------------------------------------------------------------------------
extern "C" void kernel_launch(void* const* d_in, const int* in_sizes, int n_in,
                              void* d_out, int out_size, void* d_ws, size_t ws_size,
                              hipStream_t stream) {
    (void)in_sizes; (void)n_in; (void)out_size; (void)ws_size;
    const float* x      = (const float*)d_in[0];  // cycle_curve_data [B,L,C,T]
    const float* logits = (const float*)d_in[1];  // [B,E]
    const int*   masks  = (const int*)  d_in[2];  // [B,E]
    const float* eW     = (const float*)d_in[3];  // [E,D,F]
    const float* ebias  = (const float*)d_in[4];  // [E,D]
    const float* gW     = (const float*)d_in[5];  // [G,D,F]
    const float* gbias  = (const float*)d_in[6];  // [G,D]
    float*       out    = (float*)d_out;          // [B,L,D] f32
    float*       g_ws   = (float*)d_ws;           // gates scratch [B,E]

    gates_kernel<<<1, 128, 0, stream>>>(logits, masks, g_ws);

    dim3 grid(D_ / 16, B_ / BG, L_ / (16 * LT));  // (48, 8, 2)
    moe_wmma_kernel<<<grid, NTHR, 0, stream>>>(x, eW, ebias, gW, gbias, g_ws, out);
}
